// VerticalAttention_80066780332535
// MI455X (gfx1250) — matrix operand order repaired and checked by reading, not compiled, not run on hardware
//
#include <hip/hip_runtime.h>
#include <hip/hip_bf16.h>
#include <math.h>

// VerticalAttention for MI455X (gfx1250, wave32).
// GEMMs use v_wmma_f32_16x16x32_f16 (f16 in, f32 accumulate) with A fragments
// register-resident across all column tiles; softmax/scatter path stays fp32
// with order-preserving-uint atomicMax + atomicAdd(f32).

namespace {
constexpr int kN   = 100000;  // voxels
constexpr int kM   = 12500;   // pillars
constexpr int kE   = 800000;  // edges
constexpr int kD   = 128;     // IN_DIM == EMBED
constexpr int kQKV = 384;     // 3*EMBED
constexpr unsigned kNegInfEnc = 0x007FFFFFu;  // enc(-inf)
}

typedef _Float16 v16h __attribute__((ext_vector_type(16)));
typedef _Float16 v8h  __attribute__((ext_vector_type(8)));
typedef float    v8f  __attribute__((ext_vector_type(8)));

// Order-preserving float<->uint mapping so unsigned atomicMax == float max.
__device__ __forceinline__ unsigned enc_ord(float f) {
  unsigned u = __float_as_uint(f);
  return (u & 0x80000000u) ? ~u : (u | 0x80000000u);
}
__device__ __forceinline__ float dec_ord(unsigned u) {
  return __uint_as_float((u & 0x80000000u) ? (u & 0x7FFFFFFFu) : ~u);
}

__global__ void k_fill_u32(unsigned* __restrict__ p, unsigned v, int n) {
  int i = blockIdx.x * blockDim.x + threadIdx.x;
  if (i < n) p[i] = v;
}

__global__ void k_cvt_f16(const float* __restrict__ s, _Float16* __restrict__ d, int n) {
  int i = blockIdx.x * blockDim.x + threadIdx.x;
  if (i < n) d[i] = (_Float16)s[i];
}

// WT[n*K + k] = W[k*Ncol + n]  (transpose + downconvert)
__global__ void k_cvt_f16_T(const float* __restrict__ W, _Float16* __restrict__ WT,
                            int K, int Ncol) {
  int i = blockIdx.x * blockDim.x + threadIdx.x;
  if (i >= K * Ncol) return;
  int n = i / K;
  int k = i - n * K;
  WT[i] = (_Float16)W[(size_t)k * Ncol + n];
}

// C[rows x Ncol] = A[rows x 128] * B[128 x Ncol] + bias, B given transposed as
// BT[Ncol x 128]. One wave owns a 16-row tile: A fragments (4 K-steps) stay in
// registers while the wave sweeps all 16-col tiles, streaming B from L2.
__global__ __launch_bounds__(256)
void k_gemm_wmma_f16(const _Float16* __restrict__ A,
                     const _Float16* __restrict__ BT,
                     const float* __restrict__ bias,
                     float* __restrict__ C,
                     int rowTiles, int Ncol) {
  const int lane = threadIdx.x & 31;
  const int wave = threadIdx.x >> 5;
  const int rt   = blockIdx.x * 8 + wave;   // row tile (16 rows)
  if (rt >= rowTiles) return;               // uniform per wave; EXEC stays all-ones
  const int hi  = lane >> 4;                // lane half
  const int l16 = lane & 15;

  // 16-bit A 16x32 fragment layout: lanes 0-15 hold K{0..7,16..23},
  // lanes 16-31 hold K{8..15,24..31}.  Load all 4 K-step fragments once.
  const _Float16* arow = A + (size_t)(rt * 16 + l16) * 128;
  v16h af[4];
#pragma unroll
  for (int kk = 0; kk < 4; ++kk) {
    v8h alo = *(const v8h*)(arow + kk * 32 + hi * 8);
    v8h ahi = *(const v8h*)(arow + kk * 32 + hi * 8 + 16);
#pragma unroll
    for (int i = 0; i < 8; ++i) { af[kk][i] = alo[i]; af[kk][i + 8] = ahi[i]; }
  }

  const int nTiles = Ncol >> 4;
  for (int ct = 0; ct < nTiles; ++ct) {
    // B 32x16 fragment: lane = column; lanes 0-15 hold K{0..7,16..23}, lanes 16-31 K{8..15,24..31}
    const _Float16* bcol = BT + (size_t)(ct * 16 + l16) * 128;
    if (ct + 1 < nTiles)
      __builtin_prefetch(bcol + 16 * 128, 0, 3);  // global_prefetch_b8 next strip
    v8f acc = {};
#pragma unroll
    for (int kk = 0; kk < 4; ++kk) {
      v8h blo = *(const v8h*)(bcol + kk * 32 + hi * 8);
      v8h bhi = *(const v8h*)(bcol + kk * 32 + hi * 8 + 16);
      v16h bv;
#pragma unroll
      for (int i = 0; i < 8; ++i) { bv[i] = blo[i]; bv[i + 8] = bhi[i]; }
      acc = __builtin_amdgcn_wmma_f32_16x16x32_f16(
          /*neg_a=*/false, af[kk], /*neg_b=*/false, bv,
          /*c_mod=*/(short)0, acc, /*reuse_a=*/false, /*reuse_b=*/false);
    }
    const int n = ct * 16 + l16;
    const float bb = bias[n];
#pragma unroll
    for (int r = 0; r < 8; ++r)   // C/D layout: VGPR r -> row r (+8 for upper lanes)
      C[(size_t)(rt * 16 + r + hi * 8) * Ncol + n] = acc[r] + bb;
  }
}

// attn[e] = dot(q[src], k[dst]) / sqrt(128); segment max into segmax[dst].
__global__ void k_edge_logits(const float* __restrict__ qkv,
                              const int* __restrict__ src, const int* __restrict__ dst,
                              float* __restrict__ attn, unsigned* __restrict__ segmax) {
  int g = blockIdx.x * blockDim.x + threadIdx.x;
  int e = g >> 5, lane = g & 31;
  if (e >= kE) return;
  int s = src[e], d = dst[e];
  const float4 q = *(const float4*)(qkv + (size_t)s * kQKV + lane * 4);
  const float4 k = *(const float4*)(qkv + (size_t)d * kQKV + kD + lane * 4);
  float p = q.x * k.x + q.y * k.y + q.z * k.z + q.w * k.w;
  for (int off = 16; off > 0; off >>= 1) p += __shfl_xor(p, off, 32);
  if (lane == 0) {
    float a = p * 0.08838834764831845f;  // 1/sqrt(128)
    attn[e] = a;
    atomicMax(segmax + d, enc_ord(a));
  }
}

// ebuf[e] = exp(attn - segmax[dst]); denom[dst] += ebuf[e]
__global__ void k_edge_exp(const float* __restrict__ attn, const int* __restrict__ dst,
                           const unsigned* __restrict__ segmax,
                           float* __restrict__ ebuf, float* __restrict__ denom) {
  int e = blockIdx.x * blockDim.x + threadIdx.x;
  if (e >= kE) return;
  int d = dst[e];
  float a = expf(attn[e] - dec_ord(segmax[d]));
  ebuf[e] = a;
  atomicAdd(denom + d, a);
}

// attended[dst] += (ebuf[e]/denom[dst]) * v[src]
__global__ void k_edge_scatter(const float* __restrict__ qkv,
                               const int* __restrict__ src, const int* __restrict__ dst,
                               const float* __restrict__ ebuf, const float* __restrict__ denom,
                               float* __restrict__ attended) {
  int g = blockIdx.x * blockDim.x + threadIdx.x;
  int e = g >> 5, lane = g & 31;
  if (e >= kE) return;
  int s = src[e], d = dst[e];
  float w = ebuf[e] / denom[d];
  const float4 v = *(const float4*)(qkv + (size_t)s * kQKV + 2 * kD + lane * 4);
  float* o = attended + (size_t)d * kD + lane * 4;
  atomicAdd(o + 0, w * v.x);
  atomicAdd(o + 1, w * v.y);
  atomicAdd(o + 2, w * v.z);
  atomicAdd(o + 3, w * v.w);
}

// pillar scatter: sum + max + count per pillar channel
__global__ void k_pillar(const float* __restrict__ feat, const int* __restrict__ inv,
                         float* __restrict__ psum, unsigned* __restrict__ pmax,
                         float* __restrict__ cnt) {
  int g = blockIdx.x * blockDim.x + threadIdx.x;
  int n = g >> 5, lane = g & 31;
  if (n >= kN) return;
  int m = inv[n];
  const float4 f = *(const float4*)(feat + (size_t)n * kD + lane * 4);
  size_t o = (size_t)m * kD + lane * 4;
  atomicAdd(psum + o + 0, f.x); atomicMax(pmax + o + 0, enc_ord(f.x));
  atomicAdd(psum + o + 1, f.y); atomicMax(pmax + o + 1, enc_ord(f.y));
  atomicAdd(psum + o + 2, f.z); atomicMax(pmax + o + 2, enc_ord(f.z));
  atomicAdd(psum + o + 3, f.w); atomicMax(pmax + o + 3, enc_ord(f.w));
  if (lane == 0) atomicAdd(cnt + m, 1.0f);
}

// d_out = [coor (M*3, value-cast to float), max + mean (M*128)]
__global__ void k_finalize(const int* __restrict__ coor, const float* __restrict__ psum,
                           const unsigned* __restrict__ pmax, const float* __restrict__ cnt,
                           float* __restrict__ out) {
  int i = blockIdx.x * blockDim.x + threadIdx.x;
  const int nc = kM * 3;
  if (i < nc) { out[i] = (float)coor[i]; return; }
  int j = i - nc;
  if (j >= kM * kD) return;
  int m = j >> 7;
  float c = cnt[m];
  out[i] = (c > 0.f) ? dec_ord(pmax[j]) + psum[j] / c : 0.f;
}

extern "C" void kernel_launch(void* const* d_in, const int* in_sizes, int n_in,
                              void* d_out, int out_size, void* d_ws, size_t ws_size,
                              hipStream_t stream) {
  const float* x     = (const float*)d_in[0];
  const float* W_in  = (const float*)d_in[1];
  const float* b_in  = (const float*)d_in[2];
  const float* W_out = (const float*)d_in[3];
  const float* b_out = (const float*)d_in[4];
  const int*   kmap  = (const int*)d_in[5];
  const int*   inv   = (const int*)d_in[6];
  const int*   coor  = (const int*)d_in[7];
  float* out = (float*)d_out;
  const int* src = kmap;
  const int* dst = kmap + kE;

  // Workspace carving (256B aligned)
  char* ws = (char*)d_ws;
  size_t off = 0;
  auto carve = [&](size_t bytes) -> void* {
    void* p = ws + off;
    off = (off + bytes + 255) & ~((size_t)255);
    return p;
  };
  _Float16* xh    = (_Float16*)carve((size_t)kN * kD * sizeof(_Float16));
  _Float16* WinT  = (_Float16*)carve((size_t)kQKV * kD * sizeof(_Float16));
  _Float16* WoutT = (_Float16*)carve((size_t)kD * kD * sizeof(_Float16));
  float*    qkv   = (float*)   carve((size_t)kN * kQKV * sizeof(float));
  float*    attn  = (float*)   carve((size_t)kE * sizeof(float));
  float*    ebuf  = (float*)   carve((size_t)kE * sizeof(float));
  unsigned* segmx = (unsigned*)carve((size_t)kN * sizeof(unsigned));
  float*    denom = (float*)   carve((size_t)kN * sizeof(float));
  float*    attd  = (float*)   carve((size_t)kN * kD * sizeof(float));
  _Float16* attdh = (_Float16*)carve((size_t)kN * kD * sizeof(_Float16));
  float*    out2  = (float*)   carve((size_t)kN * kD * sizeof(float));
  float*    psum  = (float*)   carve((size_t)kM * kD * sizeof(float));
  unsigned* pmax  = (unsigned*)carve((size_t)kM * kD * sizeof(unsigned));
  float*    cnt   = (float*)   carve((size_t)kM * sizeof(float));
  (void)ws_size; (void)in_sizes; (void)n_in; (void)out_size;

  const int B = 256;
  auto blocks = [](long long n, int b) { return (int)((n + b - 1) / b); };
  const int rowTiles = kN / 16;                 // 6250 (exact)
  const int gemmBlocks = (rowTiles + 7) / 8;    // 8 waves per block

  // 1) reset accumulators (harness poisons once; we must re-init every call)
  k_fill_u32<<<blocks(kN, B), B, 0, stream>>>(segmx, kNegInfEnc, kN);
  k_fill_u32<<<blocks(kN, B), B, 0, stream>>>((unsigned*)denom, 0u, kN);
  k_fill_u32<<<blocks((long long)kN * kD, B), B, 0, stream>>>((unsigned*)attd, 0u, kN * kD);
  k_fill_u32<<<blocks((long long)kM * kD, B), B, 0, stream>>>((unsigned*)psum, 0u, kM * kD);
  k_fill_u32<<<blocks((long long)kM * kD, B), B, 0, stream>>>(pmax, kNegInfEnc, kM * kD);
  k_fill_u32<<<blocks(kM, B), B, 0, stream>>>((unsigned*)cnt, 0u, kM);

  // 2) f16 conversions (weights transposed for contiguous B fragments)
  k_cvt_f16<<<blocks((long long)kN * kD, B), B, 0, stream>>>(x, xh, kN * kD);
  k_cvt_f16_T<<<blocks((long long)kD * kQKV, B), B, 0, stream>>>(W_in, WinT, kD, kQKV);
  k_cvt_f16_T<<<blocks((long long)kD * kD, B), B, 0, stream>>>(W_out, WoutT, kD, kD);

  // 3) qkv = x @ W_in + b_in   (WMMA f16->f32; A register-resident)
  k_gemm_wmma_f16<<<gemmBlocks, B, 0, stream>>>(xh, WinT, b_in, qkv, rowTiles, kQKV);

  // 4-6) edge attention: logits -> segment max -> exp/sum -> normalized scatter
  k_edge_logits<<<blocks((long long)kE * 32, B), B, 0, stream>>>(qkv, src, dst, attn, segmx);
  k_edge_exp<<<blocks(kE, B), B, 0, stream>>>(attn, dst, segmx, ebuf, denom);
  k_edge_scatter<<<blocks((long long)kE * 32, B), B, 0, stream>>>(qkv, src, dst, ebuf, denom, attd);

  // 7-8) out2 = attended @ W_out + b_out   (WMMA f16->f32)
  k_cvt_f16<<<blocks((long long)kN * kD, B), B, 0, stream>>>(attd, attdh, kN * kD);
  k_gemm_wmma_f16<<<gemmBlocks, B, 0, stream>>>(attdh, WoutT, b_out, out2, rowTiles, kD);

  // 9-10) pillar mean+max reduction, then emit (coor, out)
  k_pillar<<<blocks((long long)kN * 32, B), B, 0, stream>>>(out2, inv, psum, pmax, cnt);
  k_finalize<<<blocks((long long)kM * 3 + (long long)kM * kD, B), B, 0, stream>>>(coor, psum, pmax, cnt, out);
}
